// Discriminator_MD_42829413876116
// MI455X (gfx1250) — matrix operand order repaired and checked by reading, not compiled, not run on hardware
//
#include <hip/hip_runtime.h>
#include <hip/hip_bf16.h>

// ---------------------------------------------------------------------------
// CDNA5 (gfx1250) Discriminator + minibatch-discrimination forward pass.
// All matrix math (5 convs as implicit GEMM + 3 MD projection GEMMs) runs on
// V_WMMA_F32_16X16X32_BF16 (wave32). fp32 tensors are converted to bf16 pairs
// with single v_perm_b32 ops; accumulation is fp32 with split accumulators.
// The MD GEMM stages the streamed T tile through LDS: cooperative coalesced
// b128 global loads, double-buffered with one barrier per 32-K chunk, and
// per-lane column reads served by ds_load at immediate offsets.
// ---------------------------------------------------------------------------

typedef __attribute__((ext_vector_type(16))) __bf16    v16bf;
typedef __attribute__((ext_vector_type(8)))  unsigned  v8u;
typedef __attribute__((ext_vector_type(8)))  float     v8f;

// Pack two fp32 into a packed bf16 pair (lo in [15:0], hi in [31:16]) with one
// v_perm_b32: selector 0x07060302 takes the high halves of {S1=lo, S0=hi}.
__device__ __forceinline__ unsigned bfpack(float lo, float hi) {
  return __builtin_amdgcn_perm(__builtin_bit_cast(unsigned, hi),
                               __builtin_bit_cast(unsigned, lo), 0x07060302u);
}
__device__ __forceinline__ v16bf as_bf(v8u v) {
  return __builtin_bit_cast(v16bf, v);
}

__global__ void init_zero(float* z) {
  if (threadIdx.x < 16) z[threadIdx.x] = 0.f;
}

// ---------------------------------------------------------------------------
// Generic 2D conv (NCHW, OIHW) via implicit GEMM:
//   M = Cout tile (16), N = flattened (b,oy,ox) tile (16), K = taps x Cin.
// One wave per 16x16 output tile, 8 waves per block. KS templated so tap
// decomposition const-folds. Channel-chunk loop is unrolled x2 onto two
// accumulators (Cin/32 is even for every layer; conv1 uses only the tail).
// ---------------------------------------------------------------------------
template <int KS>
__global__ void __launch_bounds__(256)
conv2d_wmma(const float* __restrict__ in,
            const float* __restrict__ w,
            float* __restrict__ outp,
            const float* __restrict__ zerobuf,
            int Cin, int Cout, int IH, int IW,
            int OH, int OW, int strd, int pad, int Bn) {
  const int wave = threadIdx.x >> 5;
  const int lane = threadIdx.x & 31;
  const int mtiles = (Cout + 15) >> 4;
  const int Ntot   = Bn * OH * OW;
  const int ntiles = (Ntot + 15) >> 4;
  const int tile   = blockIdx.x * 8 + wave;
  if (tile >= mtiles * ntiles) return;
  const int mtile = tile % mtiles;
  const int ntile = tile / mtiles;

  const int hi = lane >> 4;
  const int lm = lane & 15;
  const int co = mtile * 16 + lm;            // A-matrix row this lane feeds
  const int n  = ntile * 16 + lm;            // B-matrix column this lane feeds
  const int ohw = OH * OW;
  const bool nvalid = (n < Ntot);
  const int nn  = nvalid ? n : 0;
  const int bb  = nn / ohw;
  const int rem = nn % ohw;
  const int oy  = rem / OW;
  const int ox  = rem % OW;
  const int coc = co < Cout ? co : Cout - 1; // clamped: garbage rows never stored
  constexpr int K2 = KS * KS;
  const int fullchunks = Cin >> 5;
  const int tailC = Cin & 31;
  const unsigned bstep = (unsigned)(IH * IW);
  const float* aptr0 = w + (size_t)coc * Cin * K2;

  v8f acc0 = {}, acc1 = {};
#pragma unroll 1
  for (int tap = 0; tap < K2; ++tap) {
    const int ky = tap / KS, kx = tap % KS;
    const int iy = oy * strd + ky - pad;
    const int ix = ox * strd + kx - pad;
    const bool pvalid = nvalid && (unsigned)iy < (unsigned)IH &&
                        (unsigned)ix < (unsigned)IW;
    // zero-stride trick: invalid pixels read zerobuf[0] with stride 0.
    const float* bp = pvalid
        ? in + (((size_t)bb * Cin) * IH + iy) * IW + ix : zerobuf;
    const unsigned bst = pvalid ? bstep : 0u;
    const float* ap = aptr0 + tap;

    auto full_chunk = [&](int cbase, v8f acc) -> v8f {
      v8u au, bu;
#pragma unroll
      for (int j = 0; j < 8; ++j) {             // A: K = hi*8 + e (+8, e>=8)
        const int e = 2 * j;
        const int c0 = cbase + hi * 8 + e + (e < 8 ? 0 : 8);
        au[j] = bfpack(ap[(size_t)c0 * K2], ap[(size_t)(c0 + 1) * K2]);
      }
#pragma unroll
      for (int j = 0; j < 8; ++j) {             // B: K = hi*8 + e (+8, e>=8)
        const unsigned c0 = (unsigned)(cbase + hi * 8 + 2 * j + (j < 4 ? 0 : 8));
        bu[j] = bfpack(bp[c0 * bst], bp[(c0 + 1) * bst]);
      }
      return __builtin_amdgcn_wmma_f32_16x16x32_bf16(
          false, as_bf(au), false, as_bf(bu), (short)0, acc, false, false);
    };

    for (int cc = 0; cc + 1 < fullchunks; cc += 2) {  // even in all layers
      acc0 = full_chunk(cc * 32, acc0);
      acc1 = full_chunk(cc * 32 + 32, acc1);
    }
    if (fullchunks & 1)                               // generality only
      acc0 = full_chunk((fullchunks - 1) * 32, acc0);

    if (tailC) {                                // ragged tail chunk
      const int cbase = fullchunks * 32;
      v8u au, bu;
#pragma unroll
      for (int j = 0; j < 8; ++j) {
        const int e = 2 * j;
        const int c0 = cbase + hi * 8 + e + (e < 8 ? 0 : 8);
        const int c1 = c0 + 1;
        const float v0 = ap[(size_t)(c0 < Cin ? c0 : Cin - 1) * K2];
        const float v1 = ap[(size_t)(c1 < Cin ? c1 : Cin - 1) * K2];
        au[j] = bfpack(c0 < Cin ? v0 : 0.f, c1 < Cin ? v1 : 0.f);
      }
#pragma unroll
      for (int j = 0; j < 8; ++j) {             // A==0 past Cin, so B garbage ok
        const int c0 = cbase + hi * 8 + 2 * j + (j < 4 ? 0 : 8);
        const int c1 = c0 + 1;
        bu[j] = bfpack(bp[(unsigned)(c0 < Cin ? c0 : Cin - 1) * bst],
                       bp[(unsigned)(c1 < Cin ? c1 : Cin - 1) * bst]);
      }
      acc0 = __builtin_amdgcn_wmma_f32_16x16x32_bf16(
          false, as_bf(au), false, as_bf(bu), (short)0, acc0, false, false);
    }
  }

  const v8f acc = acc0 + acc1;
  if (nvalid) {
#pragma unroll
    for (int r = 0; r < 8; ++r) {               // C/D: VGPR r -> row r (+8 hi)
      const int oc = mtile * 16 + (hi ? r + 8 : r);
      if (oc < Cout)
        outp[(((size_t)bb * Cout) + oc) * ohw + rem] = acc[r];
    }
  }
}

// ---------------------------------------------------------------------------
// Per-channel batch statistics (mean, population var) over (B,H,W).
// One block per channel, LDS tree reduction.
// ---------------------------------------------------------------------------
__global__ void bn_stats(const float* __restrict__ pre,
                         float* __restrict__ mean, float* __restrict__ var,
                         int C, int HW, int Bn) {
  const int c = blockIdx.x;
  float s = 0.f, s2 = 0.f;
  for (int b = 0; b < Bn; ++b) {
    const float* p = pre + ((size_t)b * C + c) * HW;
    for (int hw = threadIdx.x; hw < HW; hw += blockDim.x) {
      const float v = p[hw];
      s += v; s2 += v * v;
    }
  }
  __shared__ float sh[256], sh2[256];
  sh[threadIdx.x] = s; sh2[threadIdx.x] = s2;
  __syncthreads();
  for (int o = 128; o > 0; o >>= 1) {
    if ((int)threadIdx.x < o) {
      sh[threadIdx.x]  += sh[threadIdx.x + o];
      sh2[threadIdx.x] += sh2[threadIdx.x + o];
    }
    __syncthreads();
  }
  if (threadIdx.x == 0) {
    const float nhw = (float)(Bn * HW);
    const float m = sh[0] / nhw;
    mean[c] = m;
    var[c]  = sh2[0] / nhw - m * m;
  }
}

// ---------------------------------------------------------------------------
// BatchNorm(affine=False) + LeakyReLU(0.2); writes into the (C+md) padded
// activation slice of d_out (channel stride Ccat). C and HW are powers of 2.
// ---------------------------------------------------------------------------
__global__ void bn_lrelu(const float* __restrict__ pre,
                         const float* __restrict__ mean,
                         const float* __restrict__ var,
                         float* __restrict__ outp,
                         int cShift, int hwShift, int Ccat, int total) {
  const int idx = blockIdx.x * blockDim.x + threadIdx.x;
  if (idx >= total) return;
  const int HW = 1 << hwShift;
  const int hw = idx & (HW - 1);
  const int c  = (idx >> hwShift) & ((1 << cShift) - 1);
  const int b  = idx >> (hwShift + cShift);
  const float xn = (pre[idx] - mean[c]) * rsqrtf(var[c] + 1e-5f);
  outp[((size_t)b * Ccat + c) * HW + hw] = xn >= 0.f ? xn : 0.2f * xn;
}

// ---------------------------------------------------------------------------
// MD projection GEMM: h[64,N] = X[64,K] * T[K,N], bf16 WMMA, fp32 accum.
// Block = 8 waves = 4 row-tiles (all 64 rows) x 2 col-tiles = 64x32 panel,
// so T streams through exactly once grid-wide. The 32x32 T tile is staged in
// LDS (padded stride 36): 256 threads x one coalesced float4 global load,
// double-buffered, one barrier per 32-K chunk; per-lane column reads are
// ds_loads at immediate offsets. A loads stay per-lane float4 (b128).
// ---------------------------------------------------------------------------
__device__ __forceinline__ v8f md_lds_chunk(const float* __restrict__ arow,
                                            const float* __restrict__ rd,
                                            int k0, int hi, v8f acc) {
  v8u au, bu;
  const float4* pa0 = (const float4*)(arow + k0 + hi * 8);       // K=hi*8+e
  const float4* pa1 = (const float4*)(arow + k0 + 16 + hi * 8);  // K=16+hi*8+e
  const float4 a0 = pa0[0], a1 = pa0[1], a2 = pa1[0], a3 = pa1[1];
  au[0] = bfpack(a0.x, a0.y); au[1] = bfpack(a0.z, a0.w);
  au[2] = bfpack(a1.x, a1.y); au[3] = bfpack(a1.z, a1.w);
  au[4] = bfpack(a2.x, a2.y); au[5] = bfpack(a2.z, a2.w);
  au[6] = bfpack(a3.x, a3.y); au[7] = bfpack(a3.z, a3.w);
#pragma unroll
  for (int j = 0; j < 8; ++j)                  // B: K = hi*8 + e (+8, e>=8) from LDS
    bu[j] = bfpack(rd[(2 * j + (j < 4 ? 0 : 8)) * 36], rd[(2 * j + 1 + (j < 4 ? 0 : 8)) * 36]);
  return __builtin_amdgcn_wmma_f32_16x16x32_bf16(
      false, as_bf(au), false, as_bf(bu), (short)0, acc, false, false);
}

__global__ void __launch_bounds__(256)
md_gemm_wmma(const float* __restrict__ xact,
             const float* __restrict__ T,
             float* __restrict__ h,
             int Kdim, int Ndim, int bstride) {
  __shared__ float bt0[32 * 36];               // 32 K-rows x 32 cols, pad 4
  __shared__ float bt1[32 * 36];
  const int tid  = threadIdx.x;
  const int wave = tid >> 5;
  const int lane = tid & 31;
  const int mtile = wave & 3;                  // 4 row tiles cover 64 rows
  const int ct    = wave >> 2;                 // 2 col sub-tiles of the panel
  const int hi = lane >> 4;
  const int lm = lane & 15;
  const int row = mtile * 16 + lm;             // batch row for A
  const int n0  = blockIdx.x * 32;
  const float* arow = xact + (size_t)row * bstride;

  // cooperative tile load: 256 threads x float4 == 32x32 elements
  const int trow = tid >> 3;                   // K-row within chunk
  const int tcol = (tid & 7) * 4;              // column group
  const unsigned kstep = 32u * (unsigned)Ndim;
  unsigned goff = (unsigned)trow * (unsigned)Ndim + (unsigned)(n0 + tcol);
  float4* st0 = (float4*)(bt0 + trow * 36 + tcol);
  float4* st1 = (float4*)(bt1 + trow * 36 + tcol);
  const float* rd0 = bt0 + (hi * 8) * 36 + ct * 16 + lm;
  const float* rd1 = bt1 + (hi * 8) * 36 + ct * 16 + lm;

  // prologue: buf0 <- chunk 0, prefetch chunk 1 into registers
  float4 r = *(const float4*)(T + goff); goff += kstep;
  *st0 = r;
  r = *(const float4*)(T + goff); goff += kstep;

  v8f acc0 = {}, acc1 = {};
  for (int k0 = 0; k0 < Kdim; k0 += 64) {
    __syncthreads();                           // buf0 ready; buf1 reads drained
    *st1 = r;                                  // buf1 <- chunk k0+32
    if (k0 + 64 < Kdim) {
      r = *(const float4*)(T + goff); goff += kstep;
      __builtin_prefetch((const void*)(T + goff + kstep), 0, 0);
    }
    acc0 = md_lds_chunk(arow, rd0, k0, hi, acc0);
    __syncthreads();                           // buf1 ready; buf0 reads drained
    *st0 = r;                                  // buf0 <- chunk k0+64 (or dead)
    if (k0 + 96 < Kdim) {
      r = *(const float4*)(T + goff); goff += kstep;
    }
    acc1 = md_lds_chunk(arow, rd1, k0 + 32, hi, acc1);
  }

  const v8f acc = acc0 + acc1;
  const int ncol = n0 + ct * 16 + lm;
#pragma unroll
  for (int rI = 0; rI < 8; ++rI) {             // C/D: VGPR r -> row r (+8 hi)
    const int orow = mtile * 16 + (hi ? rI + 8 : rI);
    h[(size_t)orow * Ndim + ncol] = acc[rI];
  }
}

// ---------------------------------------------------------------------------
// MD pairwise term, faithful to the reference quirk:
//   md[i,m] = sum_{j>i} exp(-L1(h_i, h_j)) + sum_{j<i} exp(-L1(h_j, h_{j+1}))
// Writes into channel Cconv (the extra concat channel) of the output slice.
// HID = 20. h is [64, Mres*20].
// ---------------------------------------------------------------------------
__global__ void md_pairwise(const float* __restrict__ h,
                            float* __restrict__ outp,
                            int Mres, int Ccat, int Cconv) {
  const int t = blockIdx.x * blockDim.x + threadIdx.x;
  if (t >= 64 * Mres) return;
  const int i = t / Mres, m = t % Mres;
  const int Nd = Mres * 20;
  const float* hi_ = h + (size_t)i * Nd + m * 20;

  float s = 0.f;
  for (int j = 0; j < 64; ++j) {
    if (j == i) continue;
    const float* hj = h + (size_t)j * Nd + m * 20;
    float l1 = 0.f;
#pragma unroll
    for (int k = 0; k < 20; ++k) l1 += fabsf(hi_[k] - hj[k]);
    if (j > i) s += __expf(-l1);
  }
  float cum = 0.f;
  for (int j = 0; j < i; ++j) {
    const float* ha = h + (size_t)j * Nd + m * 20;
    const float* hb = ha + Nd;
    float l1 = 0.f;
#pragma unroll
    for (int k = 0; k < 20; ++k) l1 += fabsf(ha[k] - hb[k]);
    cum += __expf(-l1);
  }
  outp[((size_t)i * Ccat + Cconv) * Mres + m] = s + cum;
}

__global__ void sigmoid_head(const float* __restrict__ pre,
                             float* __restrict__ outp, int n) {
  const int t = blockIdx.x * blockDim.x + threadIdx.x;
  if (t < n) outp[t] = 1.f / (1.f + __expf(-pre[t]));
}

// ---------------------------------------------------------------------------
// Host-side orchestration.
// ---------------------------------------------------------------------------
extern "C" void kernel_launch(void* const* d_in, const int* in_sizes, int n_in,
                              void* d_out, int out_size, void* d_ws, size_t ws_size,
                              hipStream_t stream) {
  (void)in_sizes; (void)n_in; (void)out_size; (void)ws_size;
  const float* x     = (const float*)d_in[0];
  const float* w1    = (const float*)d_in[1];
  const float* w2    = (const float*)d_in[2];
  const float* w3    = (const float*)d_in[3];
  const float* w4    = (const float*)d_in[4];
  const float* w_out = (const float*)d_in[5];
  const float* T2    = (const float*)d_in[6];
  const float* T3    = (const float*)d_in[7];
  const float* T4    = (const float*)d_in[8];

  // d_out slices (concatenated tuple: out, x1, x2, x3, x4)
  float* out0 = (float*)d_out;                  // [64]
  float* x1   = out0 + 64;                      // [64, 64,32,32]
  float* x2   = x1 + (size_t)64 * 64 * 32 * 32; // [64,129,16,16]
  float* x3   = x2 + (size_t)64 * 129 * 16 * 16;// [64,257, 8, 8]
  float* x4   = x3 + (size_t)64 * 257 * 8 * 8;  // [64,513, 4, 4]

  // workspace: pre-BN buffer (16 MB max), stats, zero pad, h buffer
  float* pre   = (float*)d_ws;
  float* meanb = (float*)((char*)d_ws + (size_t)(17u << 20));
  float* varb  = meanb + 512;
  float* zbuf  = (float*)((char*)d_ws + (size_t)(17u << 20) + 8192);
  float* hbuf  = (float*)((char*)d_ws + (size_t)(18u << 20));

  const int TPB = 256;
  auto convBlocks = [](int Cout, int Ntot) {
    int tiles = ((Cout + 15) >> 4) * ((Ntot + 15) >> 4);
    return (tiles + 7) / 8;
  };

  init_zero<<<1, 32, 0, stream>>>(zbuf);

  // ---- layer 1: conv(3->64, 64x64 -> 32x32), BN+LReLU -------------------
  conv2d_wmma<5><<<convBlocks(64, 64 * 32 * 32), TPB, 0, stream>>>(
      x, w1, pre, zbuf, 3, 64, 64, 64, 32, 32, 2, 2, 64);
  bn_stats<<<64, TPB, 0, stream>>>(pre, meanb, varb, 64, 1024, 64);
  bn_lrelu<<<(64 * 64 * 1024 + TPB - 1) / TPB, TPB, 0, stream>>>(
      pre, meanb, varb, x1, 6, 10, 64, 64 * 64 * 1024);

  // ---- layer 2: conv(64->128, 32x32 -> 16x16), BN+LReLU, MD -------------
  conv2d_wmma<5><<<convBlocks(128, 64 * 16 * 16), TPB, 0, stream>>>(
      x1, w2, pre, zbuf, 64, 128, 32, 32, 16, 16, 2, 2, 64);
  bn_stats<<<128, TPB, 0, stream>>>(pre, meanb, varb, 128, 256, 64);
  bn_lrelu<<<(64 * 128 * 256 + TPB - 1) / TPB, TPB, 0, stream>>>(
      pre, meanb, varb, x2, 7, 8, 129, 64 * 128 * 256);
  md_gemm_wmma<<<5120 / 32, TPB, 0, stream>>>(x2, T2, hbuf, 32768, 5120, 129 * 256);
  md_pairwise<<<(64 * 256 + TPB - 1) / TPB, TPB, 0, stream>>>(hbuf, x2, 256, 129, 128);

  // ---- layer 3: conv(129->256, 16x16 -> 8x8), BN+LReLU, MD --------------
  conv2d_wmma<5><<<convBlocks(256, 64 * 8 * 8), TPB, 0, stream>>>(
      x2, w3, pre, zbuf, 129, 256, 16, 16, 8, 8, 2, 2, 64);
  bn_stats<<<256, TPB, 0, stream>>>(pre, meanb, varb, 256, 64, 64);
  bn_lrelu<<<(64 * 256 * 64 + TPB - 1) / TPB, TPB, 0, stream>>>(
      pre, meanb, varb, x3, 8, 6, 257, 64 * 256 * 64);
  md_gemm_wmma<<<1280 / 32, TPB, 0, stream>>>(x3, T3, hbuf, 16384, 1280, 257 * 64);
  md_pairwise<<<(64 * 64 + TPB - 1) / TPB, TPB, 0, stream>>>(hbuf, x3, 64, 257, 256);

  // ---- layer 4: conv(257->512, 8x8 -> 4x4), BN+LReLU, MD ----------------
  conv2d_wmma<5><<<convBlocks(512, 64 * 4 * 4), TPB, 0, stream>>>(
      x3, w4, pre, zbuf, 257, 512, 8, 8, 4, 4, 2, 2, 64);
  bn_stats<<<512, TPB, 0, stream>>>(pre, meanb, varb, 512, 16, 64);
  bn_lrelu<<<(64 * 512 * 16 + TPB - 1) / TPB, TPB, 0, stream>>>(
      pre, meanb, varb, x4, 9, 4, 513, 64 * 512 * 16);
  md_gemm_wmma<<<320 / 32, TPB, 0, stream>>>(x4, T4, hbuf, 8192, 320, 513 * 16);
  md_pairwise<<<(64 * 16 + TPB - 1) / TPB, TPB, 0, stream>>>(hbuf, x4, 16, 513, 512);

  // ---- head: conv(513->1, 4x4, valid) + sigmoid -------------------------
  conv2d_wmma<4><<<convBlocks(1, 64), TPB, 0, stream>>>(
      x4, w_out, pre, zbuf, 513, 1, 4, 4, 1, 1, 1, 0, 64);
  sigmoid_head<<<1, 64, 0, stream>>>(pre, out0, 64);
}
